// CausalSelfAttention_29626684408504
// MI455X (gfx1250) — matrix operand order repaired and checked by reading, not compiled, not run on hardware
//
#include <hip/hip_runtime.h>
#include <hip/hip_bf16.h>
#include <math.h>

// ---------------------------------------------------------------------------
// CDNA5 / gfx1250 causal self-attention, bf16 WMMA with f32 accumulation.
// Double-buffered LDS tiles; async global->LDS copies when the toolchain
// exposes the gfx1250 builtins (falls back to sync b128 copies otherwise).
// ---------------------------------------------------------------------------

typedef __attribute__((ext_vector_type(16))) __bf16 v16bf;
typedef __attribute__((ext_vector_type(8)))  float  v8f;
typedef __attribute__((ext_vector_type(4)))  int    v4i;

union FragU { v16bf v; uint4 q[2]; };
union Pack8 { uint4 u; __bf16 h[8]; };
union Pack4 { uint2 u; __bf16 h[4]; };

// --- async global->LDS path (gfx1250 GLOBAL_LOAD_ASYNC_TO_LDS_B128) --------
#if defined(__has_builtin)
#if __has_builtin(__builtin_amdgcn_global_load_async_to_lds_b128) && \
    __has_builtin(__builtin_amdgcn_s_wait_asynccnt)
#define ATTN_USE_ASYNC 1
#endif
#endif
#ifndef ATTN_USE_ASYNC
#define ATTN_USE_ASYNC 0
#endif

typedef __attribute__((address_space(1))) v4i global_v4i;  // AS1: __device__
typedef __attribute__((address_space(3))) v4i lds_v4i;     // AS3: __shared__

// Copy 16B global -> LDS. Async variant is tracked by ASYNCcnt and bypasses
// VGPRs (generic LDS pointers carry the LDS offset in their low 32 bits).
__device__ __forceinline__ void copy_b128(void* lds, const void* g) {
#if ATTN_USE_ASYNC
  __builtin_amdgcn_global_load_async_to_lds_b128(
      (global_v4i*)(uintptr_t)g, (lds_v4i*)(uint32_t)(uintptr_t)lds, 0, 0);
#else
  *(uint4*)lds = *(const uint4*)g;
#endif
}
__device__ __forceinline__ void wait_copies() {
#if ATTN_USE_ASYNC
  __builtin_amdgcn_s_wait_asynccnt(0);
#endif
}

// A-matrix fragment (16x32 bf16), source row-major with K contiguous.
// ISA 7.12.2: lanes 0-15 row M=lane, elems0-7 K=0..7, elems8-15 K=16..23;
// lanes 16-31 same rows, elems0-7 K=8..15, elems8-15 K=24..31.
__device__ __forceinline__ v16bf ld_frag_a(const __bf16* base, int stride, int lane) {
  const int r = lane & 15, hi = lane >> 4;
  const __bf16* p = base + (size_t)r * stride;
  FragU f;
  f.q[0] = *(const uint4*)(p + hi * 8);
  f.q[1] = *(const uint4*)(p + 16 + hi * 8);
  return f.v;
}

// B-matrix fragment (32x16 bf16): B[k][n] with storage src[n][k] row-major
// (K contiguous per column n). Lane n=lane&15, elems0-7 K=(lane>>4)*8+e, elems8-15 K=16+(lane>>4)*8+(e-8).
__device__ __forceinline__ v16bf ld_frag_b(const __bf16* base, int stride, int lane) {
  const int c = lane & 15, hi = lane >> 4;
  const __bf16* p = base + (size_t)c * stride + hi * 8;
  FragU f;
  f.q[0] = *(const uint4*)(p);
  f.q[1] = *(const uint4*)(p + 16);
  return f.v;
}

__device__ __forceinline__ v8f wmma_bf16(v16bf a, v16bf b, v8f c) {
  return __builtin_amdgcn_wmma_f32_16x16x32_bf16(false, a, false, b, (short)0, c,
                                                 false, false);
}

// ---------------------------------------------------------------------------
// fp32 -> bf16 conversion (4-wide)
// ---------------------------------------------------------------------------
__global__ void cvt_f32_bf16(const float* __restrict__ src,
                             __bf16* __restrict__ dst, int n4) {
  int i = blockIdx.x * blockDim.x + threadIdx.x;
  if (i >= n4) return;
  const float4 f = ((const float4*)src)[i];
  Pack4 p;
  p.h[0] = (__bf16)f.x; p.h[1] = (__bf16)f.y;
  p.h[2] = (__bf16)f.z; p.h[3] = (__bf16)f.w;
  ((uint2*)dst)[i] = p.u;
}

// ---------------------------------------------------------------------------
// GEMM: Out[m][n] = sum_k A[m][k] * W[n][k] + bias[n]
// A: MxK bf16 row-major, W: NxK bf16 row-major.
// Block tile 128(M) x 128(N), K-step 32, 8 waves of 32, each wave 32x64
// (2x4 WMMA tiles -> 8 WMMAs per K-step). Double-buffered LDS.
// ---------------------------------------------------------------------------
template <bool OUT_BF16>
__global__ __launch_bounds__(256) void gemm_wmma(
    const __bf16* __restrict__ A, const __bf16* __restrict__ W,
    const float* __restrict__ bias, __bf16* __restrict__ outB,
    float* __restrict__ outF, int M, int N, int K) {
  constexpr int LDT = 40;  // 32 + 8 bf16 pad, 80B rows (16B aligned)
  __shared__ __bf16 As[2][128 * LDT];
  __shared__ __bf16 Ws[2][128 * LDT];

  const int tid = threadIdx.x;
  const int lane = tid & 31, wid = tid >> 5;
  const int wm = (wid & 3) * 32;   // wave M offset within tile
  const int wn = (wid >> 2) * 64;  // wave N offset within tile
  const int m0 = blockIdx.y * 128, n0 = blockIdx.x * 128;

  v8f acc[2][4] = {};
  const int ksteps = K / 32;

  // Cooperative copy of one 128x32 A slab + 128x32 W slab into buffer `buf`.
  auto copy_tiles = [&](int kt, int buf) {
    const int k0 = kt * 32;
#pragma unroll
    for (int i = 0; i < 2; ++i) {
      const int c = tid + i * 256;
      const int row = c >> 2, kc = (c & 3) * 8;
      copy_b128(&As[buf][row * LDT + kc],
                &A[(size_t)(m0 + row) * K + k0 + kc]);
      copy_b128(&Ws[buf][row * LDT + kc],
                &W[(size_t)(n0 + row) * K + k0 + kc]);
    }
  };

  copy_tiles(0, 0);
  for (int kt = 0; kt < ksteps; ++kt) {
    wait_copies();
    __syncthreads();
    if (kt + 1 < ksteps) copy_tiles(kt + 1, (kt + 1) & 1);  // stream next slab
    const int buf = kt & 1;

    const v16bf a0 = ld_frag_a(&As[buf][(wm + 0) * LDT], LDT, lane);
    const v16bf a1 = ld_frag_a(&As[buf][(wm + 16) * LDT], LDT, lane);
#pragma unroll
    for (int ni = 0; ni < 4; ++ni) {
      const v16bf b = ld_frag_b(&Ws[buf][(wn + ni * 16) * LDT], LDT, lane);
      acc[0][ni] = wmma_bf16(a0, b, acc[0][ni]);
      acc[1][ni] = wmma_bf16(a1, b, acc[1][ni]);
    }
    __syncthreads();
  }

  // Epilogue: C/D layout -> lane (0-15) col, VGPR r -> row r (+8 for hi half)
  const int cl = lane & 15, hi = lane >> 4;
#pragma unroll
  for (int ni = 0; ni < 4; ++ni) {
    const int col = n0 + wn + ni * 16 + cl;
    const float bsv = bias[col];
#pragma unroll
    for (int mi = 0; mi < 2; ++mi) {
#pragma unroll
      for (int r = 0; r < 8; ++r) {
        const int row = m0 + wm + mi * 16 + hi * 8 + r;
        const float val = acc[mi][ni][r] + bsv;
        if constexpr (OUT_BF16)
          outB[(size_t)row * N + col] = (__bf16)val;
        else
          outF[(size_t)row * N + col] = val;
      }
    }
  }
}

// ---------------------------------------------------------------------------
// RoPE, in-place on Q and K (bf16), one thread per (b,t,h) row.
// Rotated layout: out[j] = q1*cos - q2*sin, out[j+32] = q1*sin + q2*cos
// ---------------------------------------------------------------------------
__global__ void rope_kernel(__bf16* __restrict__ Q, __bf16* __restrict__ K,
                            int BT, int H, int T, int C, int HD) {
  int idx = blockIdx.x * blockDim.x + threadIdx.x;
  if (idx >= BT * H) return;
  const int h = idx % H;
  const int bt = idx / H;
  const int t = bt % T;
  const size_t base = (size_t)bt * C + (size_t)h * HD;
  const float LOGF = 9.210340371976184f;  // ln(10000)
  __bf16* ptrs[2] = {Q + base, K + base};
#pragma unroll
  for (int p = 0; p < 2; ++p) {
    __bf16* row = ptrs[p];
    float out[64];
#pragma unroll
    for (int j = 0; j < 32; ++j) {
      const float invf = __expf(-(float)j * (LOGF / 32.0f));
      const float ang = (float)t * invf;
      float s, c;
      __sincosf(ang, &s, &c);
      const float v1 = (float)row[2 * j], v2 = (float)row[2 * j + 1];
      out[j] = v1 * c - v2 * s;
      out[j + 32] = v1 * s + v2 * c;
    }
#pragma unroll
    for (int j = 0; j < 64; ++j) row[j] = (__bf16)out[j];
  }
}

// ---------------------------------------------------------------------------
// Flash attention (causal): grid (T/64, B*H), 4 waves x 16 query rows.
// Per key tile of 64: S = Q K^T via 8 WMMAs, online softmax, O += P V via 8.
// K tile streamed async (overlaps the manual V transpose).
// ---------------------------------------------------------------------------
__global__ __launch_bounds__(128) void flash_attn(
    const __bf16* __restrict__ Q, const __bf16* __restrict__ K,
    const __bf16* __restrict__ V, __bf16* __restrict__ Y, int B, int T, int H,
    int C, int HD) {
  constexpr int LDK = 72;  // 64 + 8 pad (144B rows, 16B aligned)
  __shared__ __bf16 Ks[64 * LDK];      // [key][dim]
  __shared__ __bf16 Vs[64 * LDK];      // transposed: [dim][key]
  __shared__ __bf16 Ps[4 * 16 * LDK];  // per-wave P staging [row][key]

  const int tid = threadIdx.x, lane = tid & 31, wid = tid >> 5;
  const int qtile = blockIdx.x;
  const int b = blockIdx.y / H, h = blockIdx.y % H;
  const int q0 = qtile * 64 + wid * 16;
  const int cl = lane & 15, hi = lane >> 4;
  const size_t headoff = (size_t)h * HD;

  // Q fragments straight from global in A layout (stride C)
  const __bf16* qbase = Q + (size_t)(b * T + q0) * C + headoff;
  const v16bf qf0 = ld_frag_a(qbase, C, lane);
  const v16bf qf1 = ld_frag_a(qbase + 32, C, lane);

  v8f accO[4] = {};
  float mrow[8], lrow[8];
#pragma unroll
  for (int r = 0; r < 8; ++r) {
    mrow[r] = -1e30f;
    lrow[r] = 0.0f;
  }
  const float scale = 0.125f;  // 1/sqrt(64)

  for (int kt = 0; kt <= qtile; ++kt) {
    const int s0 = kt * 64;
    __syncthreads();
    // K tile: async global->LDS; V tile: sync load + transpose into LDS
#pragma unroll
    for (int i = 0; i < 4; ++i) {
      const int c = tid + i * 128;
      const int key = c >> 3, dc = (c & 7) * 8;
      const size_t g = (size_t)(b * T + s0 + key) * C + headoff + dc;
      copy_b128(&Ks[key * LDK + dc], &K[g]);
      Pack8 pv;
      pv.u = *(const uint4*)&V[g];
#pragma unroll
      for (int j = 0; j < 8; ++j) Vs[(dc + j) * LDK + key] = pv.h[j];
    }
    wait_copies();
    __syncthreads();

    // S = Q K^T : 16 x 64 per wave (B[k=dim][n=key] from row-major Ks)
    v8f accS[4];
#pragma unroll
    for (int st = 0; st < 4; ++st) {
      v16bf bk0 = ld_frag_b(&Ks[(st * 16) * LDK], LDK, lane);
      v16bf bk1 = ld_frag_b(&Ks[(st * 16) * LDK + 32], LDK, lane);
      v8f s = {};
      s = wmma_bf16(qf0, bk0, s);
      s = wmma_bf16(qf1, bk1, s);
      accS[st] = s;
    }

    // scale + causal mask (diagonal tile only)
    const bool lastTile = (kt == qtile);
#pragma unroll
    for (int st = 0; st < 4; ++st) {
#pragma unroll
      for (int r = 0; r < 8; ++r) {
        float v = accS[st][r] * scale;
        if (lastTile) {
          const int colg = s0 + st * 16 + cl;
          const int rowg = q0 + hi * 8 + r;
          if (colg > rowg) v = -1e30f;
        }
        accS[st][r] = v;
      }
    }

    // Online softmax: row stats live per (VGPR slot r, lane half)
#pragma unroll
    for (int r = 0; r < 8; ++r) {
      float v = fmaxf(fmaxf(accS[0][r], accS[1][r]),
                      fmaxf(accS[2][r], accS[3][r]));
      v = fmaxf(v, __shfl_xor(v, 1, 32));
      v = fmaxf(v, __shfl_xor(v, 2, 32));
      v = fmaxf(v, __shfl_xor(v, 4, 32));
      v = fmaxf(v, __shfl_xor(v, 8, 32));
      const float mnew = fmaxf(mrow[r], v);
      const float esc = __expf(mrow[r] - mnew);
      float rs = 0.0f;
#pragma unroll
      for (int st = 0; st < 4; ++st) {
        const float p = __expf(accS[st][r] - mnew);
        rs += p;
        Ps[(wid * 16 + hi * 8 + r) * LDK + st * 16 + cl] = (__bf16)p;
      }
      rs += __shfl_xor(rs, 1, 32);
      rs += __shfl_xor(rs, 2, 32);
      rs += __shfl_xor(rs, 4, 32);
      rs += __shfl_xor(rs, 8, 32);
      lrow[r] = lrow[r] * esc + rs;
      mrow[r] = mnew;
#pragma unroll
      for (int nt = 0; nt < 4; ++nt) accO[nt][r] *= esc;
    }

    // O += P V  (P reloaded from LDS in A layout; V transposed -> contiguous k)
    const v16bf pf0 = ld_frag_a(&Ps[(wid * 16) * LDK], LDK, lane);
    const v16bf pf1 = ld_frag_a(&Ps[(wid * 16) * LDK + 32], LDK, lane);
#pragma unroll
    for (int nt = 0; nt < 4; ++nt) {
      v16bf bv0 = ld_frag_b(&Vs[(nt * 16) * LDK], LDK, lane);
      v16bf bv1 = ld_frag_b(&Vs[(nt * 16) * LDK + 32], LDK, lane);
      accO[nt] = wmma_bf16(pf0, bv0, accO[nt]);
      accO[nt] = wmma_bf16(pf1, bv1, accO[nt]);
    }
  }

  // finalize: y = O / l, write bf16 back into (B,T,C) layout
#pragma unroll
  for (int r = 0; r < 8; ++r) {
    const float inv = 1.0f / lrow[r];
    const int t = q0 + hi * 8 + r;
    const size_t rowoff = (size_t)(b * T + t) * C + headoff;
#pragma unroll
    for (int nt = 0; nt < 4; ++nt)
      Y[rowoff + nt * 16 + cl] = (__bf16)(accO[nt][r] * inv);
  }
}

// ---------------------------------------------------------------------------
// Host launch
// ---------------------------------------------------------------------------
extern "C" void kernel_launch(void* const* d_in, const int* in_sizes, int n_in,
                              void* d_out, int out_size, void* d_ws,
                              size_t ws_size, hipStream_t stream) {
  (void)in_sizes; (void)n_in; (void)out_size; (void)ws_size;
  constexpr int Bc = 8, Tc = 1024, Cc = 1024, Hc = 16, HDc = 64;
  constexpr int Mc = Bc * Tc;  // 8192 rows

  const float* x  = (const float*)d_in[0];
  const float* wq = (const float*)d_in[1];
  const float* bq = (const float*)d_in[2];
  const float* wk = (const float*)d_in[3];
  const float* bk = (const float*)d_in[4];
  const float* wv = (const float*)d_in[5];
  const float* bv = (const float*)d_in[6];
  const float* wo = (const float*)d_in[7];
  const float* bo = (const float*)d_in[8];
  float* out = (float*)d_out;

  char* ws = (char*)d_ws;
  size_t off = 0;
  auto alloc = [&](size_t elems) {
    __bf16* p = (__bf16*)(ws + off);
    off += elems * sizeof(__bf16);
    return p;
  };
  __bf16* Xb  = alloc((size_t)Mc * Cc);
  __bf16* Wqb = alloc((size_t)Cc * Cc);
  __bf16* Wkb = alloc((size_t)Cc * Cc);
  __bf16* Wvb = alloc((size_t)Cc * Cc);
  __bf16* Wob = alloc((size_t)Cc * Cc);
  __bf16* Qb  = alloc((size_t)Mc * Cc);
  __bf16* Kb  = alloc((size_t)Mc * Cc);
  __bf16* Vb  = alloc((size_t)Mc * Cc);
  __bf16* Yb  = Xb;  // x no longer needed after QKV projections

  const int thr = 256;
  auto cvt = [&](const float* s, __bf16* d, size_t n) {
    int n4 = (int)(n / 4);
    cvt_f32_bf16<<<(n4 + thr - 1) / thr, thr, 0, stream>>>(s, d, n4);
  };
  cvt(x,  Xb,  (size_t)Mc * Cc);
  cvt(wq, Wqb, (size_t)Cc * Cc);
  cvt(wk, Wkb, (size_t)Cc * Cc);
  cvt(wv, Wvb, (size_t)Cc * Cc);
  cvt(wo, Wob, (size_t)Cc * Cc);

  const dim3 ggrid(Cc / 128, Mc / 128);
  gemm_wmma<true><<<ggrid, 256, 0, stream>>>(Xb, Wqb, bq, Qb, nullptr, Mc, Cc, Cc);
  gemm_wmma<true><<<ggrid, 256, 0, stream>>>(Xb, Wkb, bk, Kb, nullptr, Mc, Cc, Cc);
  gemm_wmma<true><<<ggrid, 256, 0, stream>>>(Xb, Wvb, bv, Vb, nullptr, Mc, Cc, Cc);

  rope_kernel<<<(Mc * Hc + thr - 1) / thr, thr, 0, stream>>>(Qb, Kb, Mc, Hc, Tc,
                                                             Cc, HDc);

  flash_attn<<<dim3(Tc / 64, Bc * Hc), 128, 0, stream>>>(Qb, Kb, Vb, Yb, Bc, Tc,
                                                         Hc, Cc, HDc);

  gemm_wmma<false><<<ggrid, 256, 0, stream>>>(Yb, Wob, bo, nullptr, out, Mc, Cc,
                                              Cc);
}
